// SFSConv_4277787427680
// MI455X (gfx1250) — compile-verified, matrix operand order checked by reading, not run on hardware
//
#include <hip/hip_runtime.h>

// ---------------------------------------------------------------------------
// SFSConv for MI455X (gfx1250). All GEMM-shaped stages use
// v_wmma_f32_16x16x32_bf16 (bf16 inputs, f32 accum). FFT stage is folded
// into a 32x32 circulant matrix per scale; the 33x33 depthwise Gabor conv is
// lowered to 33 Toeplitz 32x32 GEMMs per 16-row tile. Gabor input tiles are
// staged LDS via the Tensor Data Mover (tensor_load_to_lds + s_wait_tensorcnt).
// Everything L2-resident -> compute bound -> WMMA everywhere.
// ---------------------------------------------------------------------------

#define HH_   256
#define WW_   32
#define HWPIX (HH_ * WW_)      // 8192
#define BB_   8
#define PI_F  3.14159265358979323846f

typedef __attribute__((ext_vector_type(16))) __bf16 v16bf;
typedef __attribute__((ext_vector_type(8)))  float  v8f;
typedef __attribute__((ext_vector_type(4)))  unsigned u32x4;
typedef __attribute__((ext_vector_type(8)))  int      i32x8;
typedef __attribute__((ext_vector_type(4)))  int      i32x4;

union FragBF { v16bf v; unsigned short s[16]; unsigned u[8]; };

static __device__ __forceinline__ unsigned short f2bf(float f) {
    unsigned u = __float_as_uint(f);
    unsigned r = (u + 0x7FFFu + ((u >> 16) & 1u)) >> 16;
    return (unsigned short)r;
}

// ---------------------------------------------------------------------------
// Precompute: Gabor Toeplitz matrices, packed in WMMA B-fragment layout.
// Tg[o][dy][ntile][lane][v]  (dword = two packed bf16 = K pair (k0,k0+1))
//   B[k][n]: n = ntile*16 + (lane&15), k0 = (v>>2)*16 + (lane>>4)*8 + (v&3)*2
//   M[o][dy][k][n] = gabor(o, dy, kx = k - n + 16)
// ---------------------------------------------------------------------------
static __device__ float gabor_tap(int o, int dy, int kx) {
    if (kx < 0 || kx > 32) return 0.f;
    float y  = (float)(dy - 16);
    float x  = (float)(kx - 16);
    float th = (float)o * (PI_F / 4.0f);
    float ct = __cosf(th), st = __sinf(th);
    float xr =  x * ct + y * st;
    float yr = -x * st + y * ct;
    return __expf(-(xr * xr + 0.25f * yr * yr) / 32.0f) * __cosf(2.0f * PI_F * xr / 10.0f);
}

__global__ __launch_bounds__(256) void k_build_gabor(unsigned* __restrict__ Tg) {
    int idx = blockIdx.x * 256 + threadIdx.x;
    if (idx >= 4 * 33 * 2 * 32 * 8) return;
    int o  = idx / 16896;          // 16896 = 33*2*32*8
    int t  = idx % 16896;
    int dy = t >> 9;               // 512 = 2*32*8
    int r  = t & 511;
    int nt = r >> 8;
    int ln = (r >> 3) & 31;
    int v  = r & 7;
    int k0 = (v >> 2) * 16 + (ln >> 4) * 8 + (v & 3) * 2;
    int n  = nt * 16 + (ln & 15);
    float a0 = gabor_tap(o, dy, k0     - n + 16);
    float a1 = gabor_tap(o, dy, k0 + 1 - n + 16);
    Tg[idx] = (unsigned)f2bf(a0) | ((unsigned)f2bf(a1) << 16);
}

// ---------------------------------------------------------------------------
// Precompute: FrFT real circulant matrices per scale, fragment-packed.
// out[wout] = sum_j in[j] * T[s][j][wout],
// T[s][j][k] = (1/32) * sum_m cos(pi*0.8*(m/(s+1))^2 - 2*pi*m*(k-j)/32)
// ---------------------------------------------------------------------------
static __device__ float frft_elem(int s, int j, int k) {
    float inv = 1.0f / (float)(s + 1);
    float acc = 0.f;
    for (int m = 0; m < 32; ++m) {
        float t   = (float)m * inv;
        float phi = PI_F * 0.8f * t * t;
        float th  = 2.0f * PI_F * (float)(m * (k - j)) / 32.0f;
        acc += __cosf(phi - th);
    }
    return acc * (1.0f / 32.0f);
}

__global__ __launch_bounds__(256) void k_build_frft(unsigned* __restrict__ Tf) {
    int idx = blockIdx.x * 256 + threadIdx.x;
    if (idx >= 2 * 2 * 32 * 8) return;
    int v  = idx & 7;
    int ln = (idx >> 3) & 31;
    int nt = (idx >> 8) & 1;
    int s  = (idx >> 9) & 1;
    int k0 = (v >> 2) * 16 + (ln >> 4) * 8 + (v & 3) * 2;
    int n  = nt * 16 + (ln & 15);
    Tf[idx] = (unsigned)f2bf(frft_elem(s, k0, n)) |
              ((unsigned)f2bf(frft_elem(s, k0 + 1, n)) << 16);
}

// ---------------------------------------------------------------------------
// Generic pointwise (1x1) conv as WMMA GEMM:
//   out[b][co][p] = sum_ci W[co][ci] * (scale?scale[b][ci]:1) * in[b][ci][p]
// 8 waves/block: wave -> (mtile = wave % MT, ptile = wave / MT)
// ---------------------------------------------------------------------------
template <int CIN_, int COUT_, bool IN_BF16>
__global__ __launch_bounds__(256) void k_pwc(
    const float* __restrict__ Wm,          // [COUT_][CIN_]
    const void*  __restrict__ inp,         // f32 or bf16(u16) [b][CIN_][npix]
    float*       __restrict__ outp,        // [b][COUT_][npix] region
    const float* __restrict__ scale,       // optional [b][CIN_]
    int npix, long in_bstride, long out_bstride)
{
    constexpr int MT = COUT_ / 16;
    constexpr int PT = 8 / MT;
    int wave  = threadIdx.x >> 5;
    int lane  = threadIdx.x & 31;
    int mtile = wave % MT;
    int pt    = wave / MT;
    int b     = blockIdx.y;
    int p0    = (blockIdx.x * PT + pt) * 16 + (lane & 15);
    int m     = mtile * 16 + (lane & 15);
    int khalf = lane >> 4;

    const float*          inF = nullptr;
    const unsigned short* inH = nullptr;
    if constexpr (IN_BF16) inH = (const unsigned short*)inp + (size_t)b * in_bstride;
    else                   inF = (const float*)inp + (size_t)b * in_bstride;

    v8f acc = {};
    for (int kc = 0; kc < CIN_; kc += 32) {
        FragBF A, Bf;
#pragma unroll
        for (int v = 0; v < 8; ++v) {
            int k0 = kc + (v >> 2) * 16 + khalf * 8 + (v & 3) * 2;
            float w0 = Wm[m * CIN_ + k0];
            float w1 = Wm[m * CIN_ + k0 + 1];
            if (scale) {
                w0 *= scale[b * CIN_ + k0];
                w1 *= scale[b * CIN_ + k0 + 1];
            }
            A.s[2 * v]     = f2bf(w0);
            A.s[2 * v + 1] = f2bf(w1);
            if constexpr (IN_BF16) {
                Bf.s[2 * v]     = inH[(size_t)k0 * npix + p0];
                Bf.s[2 * v + 1] = inH[(size_t)(k0 + 1) * npix + p0];
            } else {
                Bf.s[2 * v]     = f2bf(inF[(size_t)k0 * npix + p0]);
                Bf.s[2 * v + 1] = f2bf(inF[(size_t)(k0 + 1) * npix + p0]);
            }
        }
        acc = __builtin_amdgcn_wmma_f32_16x16x32_bf16(
            false, A.v, false, Bf.v, (short)0, acc, false, false);
    }
    float* outB = outp + (size_t)b * out_bstride;
#pragma unroll
    for (int r = 0; r < 8; ++r)
        outB[(size_t)(mtile * 16 + r + 8 * khalf) * npix + p0] = acc[r];
}

// ---------------------------------------------------------------------------
// FrFT along W as WMMA: per block = one (b,c) image, one 16-row tile.
// 4 waves: (scale s, ntile). Writes bf16 y2[(b*64 + s*32 + c)][H][W].
// ---------------------------------------------------------------------------
__global__ __launch_bounds__(128) void k_frft(
    const float* __restrict__ fr,          // [b][32][HWPIX]
    const unsigned* __restrict__ Tf,       // [2][2][32][8]
    unsigned short* __restrict__ y2)       // bf16 [b*64+s*32+c][HWPIX]
{
    int blk   = blockIdx.x;
    int htile = blk & 15;
    int img   = blk >> 4;                  // b*32 + c
    int c = img & 31, b = img >> 5;
    int h0 = htile * 16;

    __shared__ unsigned short lds[16 * WW_];
    const float* src = fr + ((size_t)b * 32 + c) * HWPIX + h0 * WW_;
    for (int i = threadIdx.x; i < 16 * WW_; i += 128) lds[i] = f2bf(src[i]);
    __syncthreads();

    int wave = threadIdx.x >> 5, lane = threadIdx.x & 31;
    int s = wave >> 1, nt = wave & 1;
    int m = lane & 15, khalf = lane >> 4;

    FragBF A;
    const unsigned short* row = &lds[m * WW_];
#pragma unroll
    for (int v = 0; v < 8; ++v) {
        int k0 = (v >> 2) * 16 + khalf * 8 + (v & 3) * 2;
        A.s[2 * v]     = row[k0];
        A.s[2 * v + 1] = row[k0 + 1];
    }
    FragBF Bf;
    const unsigned* p = Tf + ((((s * 2) + nt) * 32 + lane) << 3);
#pragma unroll
    for (int i = 0; i < 8; ++i) Bf.u[i] = p[i];

    v8f acc = {};
    acc = __builtin_amdgcn_wmma_f32_16x16x32_bf16(
        false, A.v, false, Bf.v, (short)0, acc, false, false);

    unsigned short* dst = y2 + ((size_t)b * 64 + s * 32 + c) * HWPIX;
    int n = nt * 16 + (lane & 15);
#pragma unroll
    for (int r = 0; r < 8; ++r)
        dst[(h0 + r + 8 * khalf) * WW_ + n] = f2bf(acc[r]);
}

// ---------------------------------------------------------------------------
// Depthwise 33x33 Gabor conv lowered to 33 Toeplitz GEMMs per row-tile.
// Block = one (b,s,c) image x one 16-row tile. The in-range part of the
// 48-row input tile is staged into LDS by the Tensor Data Mover (one 2-D
// D# descriptor: 32-wide rows, 2-byte elements); halo rows are zero-filled
// with ds stores in a disjoint LDS region. Wave 0 issues the TDM op and
// waits on TENSORcnt; the workgroup barrier releases the other waves.
// 8 waves: (orientation o, ntile). 33 WMMAs per wave.
// Output bf16 g[b][s][o][c][H][W]  (channel order matches fpu_pwc transpose).
// ---------------------------------------------------------------------------
__global__ __launch_bounds__(256) void k_gabor(
    const unsigned short* __restrict__ y2, // bf16 [img][HWPIX]
    const unsigned* __restrict__ Tg,       // [4][33][2][32][8]
    unsigned short* __restrict__ g)        // bf16 [b][2][4][32][HWPIX]
{
    int blk   = blockIdx.x;
    int htile = blk & 15;
    int img   = blk >> 4;                  // b*64 + s*32 + c
    int h0    = htile * 16;

    __shared__ unsigned short lds[48 * WW_];
    const unsigned short* src = y2 + (size_t)img * HWPIX;

    int lo    = h0 - 16;                   // desired rows [lo, lo+48)
    int start = lo < 0 ? 0 : lo;           // clamped valid range [start, end)
    int end   = (h0 + 32) > HH_ ? HH_ : (h0 + 32);
    int nrows = end - start;               // 32..48 (block-uniform)

    // zero-fill halo rows (disjoint from the TDM destination region)
    for (int i = threadIdx.x; i < 48 * WW_; i += 256) {
        int hr = lo + (i >> 5);
        if (hr < 0 || hr >= HH_) lds[i] = 0;
    }

    if ((threadIdx.x >> 5) == 0) {
        // --- Tensor DMA descriptor (D#), groups 0/1; groups 2/3 zero (2-D) ---
        unsigned long long ga =
            (unsigned long long)(uintptr_t)(src + (size_t)start * WW_);
        unsigned lds_byte =
            (unsigned)(uintptr_t)(&lds[0]) + (unsigned)(start - lo) * (WW_ * 2);
        u32x4 g0;
        g0[0] = 1u;                                        // count=1, user mode
        g0[1] = lds_byte;                                  // lds_addr (bytes)
        g0[2] = (unsigned)(ga & 0xFFFFFFFFull);            // global_addr lo
        g0[3] = (unsigned)((ga >> 32) & 0x1FFFFFFull) | (2u << 30); // hi|type=2
        i32x8 g1;
        g1[0] = (int)(1u << 16);                           // data_size=1 (2B)
        g1[1] = (int)((unsigned)WW_ << 16);                // tensor_dim0 = 32
        g1[2] = (int)(((unsigned)nrows & 0xFFFFu) << 16);  // tensor_dim1 = nrows
        g1[3] = (int)((unsigned)WW_ << 16);                // tile_dim0 = 32
        g1[4] = (int)((unsigned)nrows & 0xFFFFu);          // tile_dim1 = nrows
        g1[5] = (int)WW_;                                  // dim0 stride = 32
        g1[6] = 0;
        g1[7] = 0;
        i32x4 z4 = { 0, 0, 0, 0 };
        i32x8 z8 = { 0, 0, 0, 0, 0, 0, 0, 0 };
        __builtin_amdgcn_tensor_load_to_lds(g0, g1, z4, z4, z8, 0);
        __builtin_amdgcn_s_wait_tensorcnt(0);
    }
    __syncthreads();

    int wave = threadIdx.x >> 5, lane = threadIdx.x & 31;
    int o = wave >> 1, nt = wave & 1;
    int m = lane & 15, khalf = lane >> 4;

    v8f acc = {};
    for (int dy = 0; dy < 33; ++dy) {
        FragBF A;
        const unsigned short* row = &lds[(m + dy) * WW_];
#pragma unroll
        for (int v = 0; v < 8; ++v) {
            int k0 = (v >> 2) * 16 + khalf * 8 + (v & 3) * 2;
            A.s[2 * v]     = row[k0];
            A.s[2 * v + 1] = row[k0 + 1];
        }
        FragBF Bf;
        const unsigned* p = Tg + ((size_t)(((o * 33 + dy) * 2 + nt) * 32 + lane) << 3);
#pragma unroll
        for (int i = 0; i < 8; ++i) Bf.u[i] = p[i];
        acc = __builtin_amdgcn_wmma_f32_16x16x32_bf16(
            false, A.v, false, Bf.v, (short)0, acc, false, false);
    }

    int c = img & 31, s = (img >> 5) & 1, b = img >> 6;
    size_t base = ((((size_t)b * 2 + s) * 4 + o) * 32 + c) * (size_t)HWPIX;
    int n = nt * 16 + (lane & 15);
#pragma unroll
    for (int r = 0; r < 8; ++r)
        g[base + (size_t)(h0 + r + 8 * khalf) * WW_ + n] = f2bf(acc[r]);
}

// ---------------------------------------------------------------------------
// Spatial pyramid: thread = (b, oc in 0..7, h, w) computes all 4 group convs
// (3/5/7/9) and writes cumulative sums o0..o3 (1.4 GFLOP -> plain VALU).
// ---------------------------------------------------------------------------
__global__ __launch_bounds__(256) void k_pyramid(
    const float* __restrict__ sp,          // [b][32][HWPIX]
    const float* __restrict__ w3, const float* __restrict__ w5,
    const float* __restrict__ w7, const float* __restrict__ w9,
    float* __restrict__ o)                 // [b][32][HWPIX] cumulative
{
    int idx = blockIdx.x * 256 + threadIdx.x;
    if (idx >= BB_ * 8 * HWPIX) return;
    int w_ = idx & 31;
    int h  = (idx >> 5) & 255;
    int oc = (idx >> 13) & 7;
    int b  = idx >> 16;

    const float* in_b = sp + (size_t)b * 32 * HWPIX;
    const float* wk[4] = { w3, w5, w7, w9 };
    const int    ks[4] = { 3, 5, 7, 9 };
    float accum = 0.f;
    for (int gidx = 0; gidx < 4; ++gidx) {
        int K = ks[gidx], pad = K >> 1;
        const float* wg = wk[gidx];
        float p = 0.f;
        for (int ic = 0; ic < 8; ++ic) {
            const float* im   = in_b + (size_t)(gidx * 8 + ic) * HWPIX;
            const float* wrow = wg + (size_t)(oc * 8 + ic) * K * K;
            for (int ky = 0; ky < K; ++ky) {
                int hh = h + ky - pad;
                if (hh < 0 || hh >= HH_) continue;
                for (int kx = 0; kx < K; ++kx) {
                    int xx = w_ + kx - pad;
                    if (xx < 0 || xx >= WW_) continue;
                    p += im[hh * WW_ + xx] * wrow[ky * K + kx];
                }
            }
        }
        accum += p;
        o[((size_t)b * 32 + gidx * 8 + oc) * HWPIX + h * WW_ + w_] = accum;
    }
}

// ---------------------------------------------------------------------------
// Channel means over H*W, then per-batch softmax over 64 channels.
// ---------------------------------------------------------------------------
__global__ __launch_bounds__(256) void k_mean(const float* __restrict__ z,
                                              float* __restrict__ s) {
    int bc = blockIdx.x;                   // b*64 + c
    const float* p = z + (size_t)bc * HWPIX;
    float acc = 0.f;
    for (int i = threadIdx.x; i < HWPIX; i += 256) acc += p[i];
    __shared__ float red[256];
    red[threadIdx.x] = acc;
    __syncthreads();
    for (int st = 128; st > 0; st >>= 1) {
        if (threadIdx.x < st) red[threadIdx.x] += red[threadIdx.x + st];
        __syncthreads();
    }
    if (threadIdx.x == 0) s[bc] = red[0] * (1.0f / (float)HWPIX);
}

__global__ __launch_bounds__(64) void k_softmax(const float* __restrict__ s,
                                                float* __restrict__ att) {
    int b = blockIdx.x, t = threadIdx.x;   // 64 threads
    __shared__ float v[64];
    __shared__ float mx, sum;
    v[t] = s[b * 64 + t];
    __syncthreads();
    if (t == 0) {
        float m = v[0];
        for (int i = 1; i < 64; ++i) m = fmaxf(m, v[i]);
        mx = m;
    }
    __syncthreads();
    float e = __expf(v[t] - mx);
    v[t] = e;
    __syncthreads();
    if (t == 0) {
        float sm = 0.f;
        for (int i = 0; i < 64; ++i) sm += v[i];
        sum = sm;
    }
    __syncthreads();
    att[b * 64 + t] = e / sum;
}

// ---------------------------------------------------------------------------
extern "C" void kernel_launch(void* const* d_in, const int* in_sizes, int n_in,
                              void* d_out, int out_size, void* d_ws, size_t ws_size,
                              hipStream_t stream) {
    (void)in_sizes; (void)n_in; (void)out_size; (void)ws_size;
    const float* x      = (const float*)d_in[0];
    const float* w_sp   = (const float*)d_in[1];
    const float* w_fr   = (const float*)d_in[2];
    const float* w_s3   = (const float*)d_in[3];
    const float* w_s5   = (const float*)d_in[4];
    const float* w_s7   = (const float*)d_in[5];
    const float* w_s9   = (const float*)d_in[6];
    const float* w_fin  = (const float*)d_in[7];
    const float* w_fpu  = (const float*)d_in[8];
    const float* w_fus  = (const float*)d_in[9];
    float* out = (float*)d_out;

    char* ws = (char*)d_ws;
    size_t off = 0;
    auto alloc = [&](size_t bytes) { char* p = ws + off; off = (off + bytes + 255) & ~(size_t)255; return p; };
    unsigned*       Tg    = (unsigned*)alloc(4u * 33u * 2u * 32u * 8u * 4u);      // 270 KB
    unsigned*       Tf    = (unsigned*)alloc(2u * 2u * 32u * 8u * 4u);            // 4 KB
    float*          spb   = (float*)alloc((size_t)BB_ * 32 * HWPIX * 4);          // 8 MB
    float*          frb   = (float*)alloc((size_t)BB_ * 32 * HWPIX * 4);          // 8 MB
    float*          ob    = (float*)alloc((size_t)BB_ * 32 * HWPIX * 4);          // 8 MB
    unsigned short* y2    = (unsigned short*)alloc((size_t)BB_ * 64 * HWPIX * 2); // 8 MB
    unsigned short* gbuf  = (unsigned short*)alloc((size_t)BB_ * 256 * HWPIX * 2);// 32 MB
    float*          zbuf  = (float*)alloc((size_t)BB_ * 64 * HWPIX * 4);          // 16 MB
    float*          sbuf  = (float*)alloc((size_t)BB_ * 64 * 4);
    float*          attb  = (float*)alloc((size_t)BB_ * 64 * 4);

    // 0) constant matrices (Gabor Toeplitz + FrFT circulants), frag-packed bf16
    k_build_gabor<<<(4 * 33 * 2 * 32 * 8 + 255) / 256, 256, 0, stream>>>(Tg);
    k_build_frft<<<4, 256, 0, stream>>>(Tf);

    // 1) channel pointwise convs on the split input
    k_pwc<32, 32, false><<<dim3(HWPIX / 64, BB_), 256, 0, stream>>>(
        w_sp, x, spb, nullptr, HWPIX, 64L * HWPIX, 32L * HWPIX);
    k_pwc<32, 32, false><<<dim3(HWPIX / 64, BB_), 256, 0, stream>>>(
        w_fr, x + 32 * HWPIX, frb, nullptr, HWPIX, 64L * HWPIX, 32L * HWPIX);

    // 2) spatial pyramid (direct) + final 1x1 -> y_s (zbuf channels 0..31)
    k_pyramid<<<(BB_ * 8 * HWPIX + 255) / 256, 256, 0, stream>>>(
        spb, w_s3, w_s5, w_s7, w_s9, ob);
    k_pwc<32, 32, false><<<dim3(HWPIX / 64, BB_), 256, 0, stream>>>(
        w_fin, ob, zbuf, nullptr, HWPIX, 32L * HWPIX, 64L * HWPIX);

    // 3) FrFT (circulant GEMM) -> Gabor (TDM-staged Toeplitz GEMMs) -> y_f
    k_frft<<<BB_ * 32 * 16, 128, 0, stream>>>(frb, Tf, y2);
    k_gabor<<<BB_ * 64 * 16, 256, 0, stream>>>(y2, Tg, gbuf);
    k_pwc<256, 32, true><<<dim3(HWPIX / 64, BB_), 256, 0, stream>>>(
        w_fpu, gbuf, zbuf + 32L * HWPIX, nullptr, HWPIX, 256L * HWPIX, 64L * HWPIX);

    // 4) attention (means + softmax), then fusion GEMM with attention folded
    //    into the per-batch weight matrix.
    k_mean<<<BB_ * 64, 256, 0, stream>>>(zbuf, sbuf);
    k_softmax<<<BB_, 64, 0, stream>>>(sbuf, attb);
    k_pwc<64, 64, false><<<dim3(HWPIX / 32, BB_), 256, 0, stream>>>(
        w_fus, zbuf, out, attb, HWPIX, 64L * HWPIX, 64L * HWPIX);
}
